// AttentionBasedModel_137438954267
// MI455X (gfx1250) — matrix operand order repaired and checked by reading, not compiled, not run on hardware
//
#include <hip/hip_runtime.h>
#include <hip/hip_bf16.h>

typedef __bf16 bf16;
typedef __attribute__((ext_vector_type(8)))  __bf16 v8bf;
typedef __attribute__((ext_vector_type(16))) __bf16 v16bf;
typedef __attribute__((ext_vector_type(8)))  float  v8f;

#define B_   32
#define S_   1024
#define D_   512
#define H_   8
#define L_   3
#define FF_  2048
#define C_   6
#define DH_  64
#define QKV3 (3*D_)   /* 1536 */

// ---------------------------------------------------------------------------
// CDNA5 async copy: global -> LDS (ASYNCcnt-tracked, bypasses VGPRs)
// ---------------------------------------------------------------------------
__device__ __forceinline__ void async_cp16(unsigned lds_off, unsigned gbyte_off,
                                           const void* sbase) {
  asm volatile("global_load_async_to_lds_b128 %0, %1, %2"
               :: "v"(lds_off), "v"(gbyte_off), "s"(sbase)
               : "memory");
}
__device__ __forceinline__ void wait_async0() {
  asm volatile("s_wait_asynccnt 0x0" ::: "memory");
}

// ---------------------------------------------------------------------------
// VALU 16-lane butterfly max via v_permlane16_b32 (no LDS round-trip)
// ---------------------------------------------------------------------------
__device__ __forceinline__ float xormax16(float v) {
  int x, y;
  x = __float_as_int(v);
  y = __builtin_amdgcn_permlane16(x, x, 0x67452301u, 0xEFCDAB89u, false, false); // ^1
  v = fmaxf(v, __int_as_float(y)); x = __float_as_int(v);
  y = __builtin_amdgcn_permlane16(x, x, 0x54761032u, 0xDCFE98BAu, false, false); // ^2
  v = fmaxf(v, __int_as_float(y)); x = __float_as_int(v);
  y = __builtin_amdgcn_permlane16(x, x, 0x32107654u, 0xBA98FEDCu, false, false); // ^4
  v = fmaxf(v, __int_as_float(y)); x = __float_as_int(v);
  y = __builtin_amdgcn_permlane16(x, x, 0xFEDCBA98u, 0x76543210u, false, false); // ^8
  v = fmaxf(v, __int_as_float(y));
  return v;
}

// ---------------------------------------------------------------------------
// WMMA helpers (CDNA5 wave32, 16x16x32 bf16 -> f32)
// ---------------------------------------------------------------------------
__device__ __forceinline__ v16bf cat8(v8bf lo, v8bf hi) {
  v16bf r;
#pragma unroll
  for (int i = 0; i < 8; ++i) { r[i] = lo[i]; r[i + 8] = hi[i]; }
  return r;
}

__device__ __forceinline__ v8f wmma_bf16(v16bf a, v16bf b, v8f c) {
  return __builtin_amdgcn_wmma_f32_16x16x32_bf16(
      false, a, false, b, (short)0, c, false, false);
}

// A fragment (16x32 bf16, row-major tile)
__device__ __forceinline__ v16bf load_a_frag(const bf16* base, int stride, int lane) {
  int half = lane >> 4, m = lane & 15;
  const bf16* p = base + m * stride + 8 * half;
  v8bf lo = *(const v8bf*)(p);
  v8bf hi = *(const v8bf*)(p + 16);
  return cat8(lo, hi);
}

// B fragment (32x16 bf16) from an N-major tile Bt[n][k]
__device__ __forceinline__ v16bf load_b_frag(const bf16* base, int stride, int lane) {
  int half = lane >> 4, n = lane & 15;
  const bf16* p = base + n * stride + 8 * half;
  v8bf lo = *(const v8bf*)(p);
  v8bf hi = *(const v8bf*)(p + 16);
  return cat8(lo, hi);
}

// ---------------------------------------------------------------------------
// Weight convert: fp32 [K][N] -> bf16 transposed [N][K]
// ---------------------------------------------------------------------------
__global__ void convw_kernel(const float* __restrict__ qkv_w, const float* __restrict__ fc_w,
                             bf16* __restrict__ qkvwT, bf16* __restrict__ fcwT) {
  size_t idx = (size_t)blockIdx.x * 256 + threadIdx.x;
  const size_t T1 = (size_t)L_ * QKV3 * D_;
  const size_t T2 = (size_t)L_ * D_ * D_;
  if (idx < T1) {
    size_t l = idx / ((size_t)QKV3 * D_);
    size_t r = idx % ((size_t)QKV3 * D_);
    size_t n = r / D_, k = r % D_;
    qkvwT[idx] = (bf16)qkv_w[(l * D_ + k) * QKV3 + n];
  } else if (idx < T1 + T2) {
    size_t i2 = idx - T1;
    size_t l = i2 / ((size_t)D_ * D_);
    size_t r = i2 % ((size_t)D_ * D_);
    size_t n = r / D_, k = r % D_;
    fcwT[i2] = (bf16)fc_w[(l * D_ + k) * D_ + n];
  }
}

// ---------------------------------------------------------------------------
// Embedding + sinusoidal positional encoding
// ---------------------------------------------------------------------------
__global__ void embed_kernel(const int* __restrict__ tokens, const float* __restrict__ emb,
                             float* __restrict__ x, bf16* __restrict__ xh) {
  size_t idx = (size_t)blockIdx.x * 256 + threadIdx.x;
  int d = (int)(idx & (D_ - 1));
  size_t bs = idx >> 9;
  int s = (int)(bs & (S_ - 1));
  int tok = tokens[bs];
  float v = emb[(size_t)tok * D_ + d];
  float freq = __expf(-(float)(d & ~1) * (9.210340371976184f / (float)D_));
  float ang = (float)s * freq;
  v += (d & 1) ? __cosf(ang) : __sinf(ang);
  x[idx] = v;
  xh[idx] = (bf16)v;
}

// ---------------------------------------------------------------------------
// Tiled WMMA GEMM, double-buffered async pipeline.
// out[M,N] = A[M,K](bf16) @ Bt[N,K](bf16)^T + bias
// 256 threads = 8 waves; block tile 128x128, BK=64; wave tile 32x64.
// BF16_OUT instantiation is the qkv GEMM: it also pre-scales the Q columns
// by dh^-0.5 so attention softmax needs no per-element scale.
// ---------------------------------------------------------------------------
#define GST 72
template <bool BF16_OUT>
__global__ __launch_bounds__(256) void gemm_kernel(
    const bf16* __restrict__ A, const bf16* __restrict__ Bt,
    const float* __restrict__ bias, void* __restrict__ out,
    int M, int N, int K) {
  __shared__ bf16 As[2][128 * GST];
  __shared__ bf16 Bs[2][128 * GST];
  const int tid = threadIdx.x;
  const int lane = tid & 31, wave = tid >> 5;
  const int wm = wave & 3, wn = wave >> 2;
  const int half = lane >> 4, ln = lane & 15;
  const int bm = blockIdx.x * 128, bn = blockIdx.y * 128;

  auto load_tiles = [&](int kt, int buf) {
#pragma unroll
    for (int i = 0; i < 4; ++i) {
      int c = tid + 256 * i;
      int row = c >> 3, seg = c & 7;
      async_cp16((unsigned)(size_t)(void*)&As[buf][row * GST + seg * 8],
                 (unsigned)(((size_t)(bm + row) * K + kt + seg * 8) * sizeof(bf16)), A);
      async_cp16((unsigned)(size_t)(void*)&Bs[buf][row * GST + seg * 8],
                 (unsigned)(((size_t)(bn + row) * K + kt + seg * 8) * sizeof(bf16)), Bt);
    }
  };

  v8f acc[2][4] = {};
  const int nt = K / 64;
  load_tiles(0, 0);
  wait_async0();
  __syncthreads();
  for (int t = 0; t < nt; ++t) {
    int buf = t & 1;
    if (t + 1 < nt) load_tiles((t + 1) * 64, buf ^ 1);  // prefetch next tile
#pragma unroll
    for (int kc = 0; kc < 2; ++kc) {
      v16bf af[2], bfm[4];
#pragma unroll
      for (int i = 0; i < 2; ++i)
        af[i] = load_a_frag(&As[buf][(wm * 32 + i * 16) * GST + kc * 32], GST, lane);
#pragma unroll
      for (int j = 0; j < 4; ++j)
        bfm[j] = load_b_frag(&Bs[buf][(wn * 64 + j * 16) * GST + kc * 32], GST, lane);
#pragma unroll
      for (int i = 0; i < 2; ++i)
#pragma unroll
        for (int j = 0; j < 4; ++j)
          acc[i][j] = wmma_bf16(af[i], bfm[j], acc[i][j]);
    }
    wait_async0();
    __syncthreads();
  }
#pragma unroll
  for (int j = 0; j < 4; ++j) {
    int col = bn + wn * 64 + j * 16 + ln;
    float bv = bias[col];
    float mul = 1.f;
    if constexpr (BF16_OUT) {          // qkv GEMM: pre-scale Q columns
      mul = ((col % (3 * DH_)) < DH_) ? 0.125f : 1.f;
    }
#pragma unroll
    for (int i = 0; i < 2; ++i) {
#pragma unroll
      for (int r = 0; r < 8; ++r) {
        int row = bm + wm * 32 + i * 16 + r + 8 * half;
        float v = (acc[i][j][r] + bv) * mul;
        size_t o = (size_t)row * N + col;
        if constexpr (BF16_OUT) ((bf16*)out)[o] = (bf16)v;
        else                    ((float*)out)[o] = v;
      }
    }
  }
}

// ---------------------------------------------------------------------------
// Flash attention: grid (B*H, S/64), 128 threads = 4 waves; wave = 16 q rows.
// Q is pre-scaled by dh^-0.5 in the qkv GEMM. Softmax denominator via the
// P x ones WMMA trick; row max via v_permlane16 VALU butterflies.
// ---------------------------------------------------------------------------
#define KST 72
__global__ __launch_bounds__(128) void attn_kernel(const bf16* __restrict__ qkv,
                                                   bf16* __restrict__ o) {
  __shared__ bf16 Ks[64 * KST];       // K tile [key][dh]
  __shared__ bf16 Vt[64 * KST];       // V tile transposed [dh][key]
  __shared__ bf16 Ps[4 * 16 * KST];   // per-wave P scratch [qrow][key]
  const int b = blockIdx.x >> 3, h = blockIdx.x & 7;
  const int qt = blockIdx.y;
  const int tid = threadIdx.x, lane = tid & 31, wave = tid >> 5;
  const int half = lane >> 4, ln = lane & 15;
  const int s0 = qt * 64 + wave * 16;
  const size_t qkvBase = (size_t)b * S_ * QKV3 + (size_t)h * (3 * DH_);

  v16bf onesb;
#pragma unroll
  for (int i = 0; i < 16; ++i) onesb[i] = (bf16)1.0f;

  // Q fragments kept in registers for the whole key loop
  v16bf qa[2];
  {
    const bf16* qrow = qkv + qkvBase + (size_t)(s0 + ln) * QKV3;
#pragma unroll
    for (int c = 0; c < 2; ++c) {
      v8bf lo = *(const v8bf*)(qrow + c * 32 + 8 * half);
      v8bf hi = *(const v8bf*)(qrow + c * 32 + 16 + 8 * half);
      qa[c] = cat8(lo, hi);
    }
  }

  v8f oacc[4] = {};
  v8f lacc = {};              // running softmax denominator (P x ones)
  float mrow[8];
#pragma unroll
  for (int r = 0; r < 8; ++r) mrow[r] = -1e30f;

  for (int kt = 0; kt < S_ / 64; ++kt) {
    __syncthreads();
    // 1) issue all async K-tile copies (global->LDS, no VGPRs)
#pragma unroll
    for (int i = 0; i < 4; ++i) {
      int c = tid + 128 * i;
      int row = c >> 3, seg = c & 7;
      unsigned loff = (unsigned)(size_t)(void*)&Ks[row * KST + seg * 8];
      unsigned goff = (unsigned)((qkvBase + (size_t)(kt * 64 + row) * QKV3 + DH_ + seg * 8) *
                                 sizeof(bf16));
      async_cp16(loff, goff, qkv);
    }
    // 2) issue all V-tile global loads (4 in flight), then scatter-transpose
    v8bf vv[4];
#pragma unroll
    for (int i = 0; i < 4; ++i) {
      int c = tid + 128 * i;
      int row = c >> 3, seg = c & 7;
      vv[i] = *(const v8bf*)(qkv + qkvBase + (size_t)(kt * 64 + row) * QKV3 + 2 * DH_ + seg * 8);
    }
#pragma unroll
    for (int i = 0; i < 4; ++i) {
      int c = tid + 128 * i;
      int row = c >> 3, seg = c & 7;
#pragma unroll
      for (int j = 0; j < 8; ++j) Vt[(seg * 8 + j) * KST + row] = vv[i][j];
    }
    wait_async0();
    __syncthreads();

    // Scores S = Q K^T (16 x 64 per wave); Q already carries dh^-0.5
    v8f sacc[4];
#pragma unroll
    for (int nf = 0; nf < 4; ++nf) {
      v8f z = {};
      v16bf kb0 = load_b_frag(&Ks[(nf * 16) * KST + 0],  KST, lane);
      v16bf kb1 = load_b_frag(&Ks[(nf * 16) * KST + 32], KST, lane);
      z = wmma_bf16(qa[0], kb0, z);
      z = wmma_bf16(qa[1], kb1, z);
      sacc[nf] = z;
    }

    // Online softmax: row max via VALU permlane butterflies
#pragma unroll
    for (int r = 0; r < 8; ++r) {
      float v = fmaxf(fmaxf(sacc[0][r], sacc[1][r]), fmaxf(sacc[2][r], sacc[3][r]));
      v = xormax16(v);
      float mnew = fmaxf(mrow[r], v);
      float alpha = __expf(mrow[r] - mnew);
      mrow[r] = mnew;
      lacc[r] *= alpha;
#pragma unroll
      for (int od = 0; od < 4; ++od) oacc[od][r] *= alpha;
#pragma unroll
      for (int nf = 0; nf < 4; ++nf)
        sacc[nf][r] = __expf(sacc[nf][r] - mnew);
    }

    // Re-fragment P via per-wave LDS (same-wave DS ops are in-order)
    bf16* Pw = &Ps[wave * 16 * KST];
#pragma unroll
    for (int nf = 0; nf < 4; ++nf)
#pragma unroll
      for (int r = 0; r < 8; ++r)
        Pw[(r + 8 * half) * KST + nf * 16 + ln] = (bf16)sacc[nf][r];

    v16bf pa[2];
#pragma unroll
    for (int c = 0; c < 2; ++c) {
      const bf16* pr = &Pw[ln * KST + c * 32];
      v8bf lo = *(const v8bf*)(pr + 8 * half);
      v8bf hi = *(const v8bf*)(pr + 16 + 8 * half);
      pa[c] = cat8(lo, hi);
    }
    // Denominator: lacc += P x ones (every lane gets its rows' sum)
    lacc = wmma_bf16(pa[0], onesb, lacc);
    lacc = wmma_bf16(pa[1], onesb, lacc);
#pragma unroll
    for (int od = 0; od < 4; ++od) {
      v16bf vb0 = load_b_frag(&Vt[(od * 16) * KST + 0],  KST, lane);
      v16bf vb1 = load_b_frag(&Vt[(od * 16) * KST + 32], KST, lane);
      oacc[od] = wmma_bf16(pa[0], vb0, oacc[od]);
      oacc[od] = wmma_bf16(pa[1], vb1, oacc[od]);
    }
  }

  float inv[8];
#pragma unroll
  for (int r = 0; r < 8; ++r) inv[r] = 1.f / lacc[r];
#pragma unroll
  for (int od = 0; od < 4; ++od) {
#pragma unroll
    for (int r = 0; r < 8; ++r) {
      int s = s0 + r + 8 * half;
      int d = h * DH_ + od * 16 + ln;
      o[((size_t)(b * S_ + s)) * D_ + d] = (bf16)(oacc[od][r] * inv[r]);
    }
  }
}

// ---------------------------------------------------------------------------
// Residual + LayerNorm: x = LN(proj + x); writes fp32 x and bf16 mirror
// ---------------------------------------------------------------------------
__global__ __launch_bounds__(256) void ln_kernel(
    const float* __restrict__ proj, const float* __restrict__ xin,
    const float* __restrict__ gamma, const float* __restrict__ beta,
    float* __restrict__ xout, bf16* __restrict__ xh) {
  __shared__ float red[256];
  const int row = blockIdx.x, tid = threadIdx.x;
  size_t base = (size_t)row * D_;
  float v0 = proj[base + tid] + xin[base + tid];
  float v1 = proj[base + tid + 256] + xin[base + tid + 256];
  red[tid] = v0 + v1;
  __syncthreads();
  for (int s = 128; s > 0; s >>= 1) { if (tid < s) red[tid] += red[tid + s]; __syncthreads(); }
  float mean = red[0] * (1.f / 512.f);
  __syncthreads();
  float d0 = v0 - mean, d1 = v1 - mean;
  red[tid] = d0 * d0 + d1 * d1;
  __syncthreads();
  for (int s = 128; s > 0; s >>= 1) { if (tid < s) red[tid] += red[tid + s]; __syncthreads(); }
  float rstd = rsqrtf(red[0] * (1.f / 512.f) + 1e-5f);
  float y0 = d0 * rstd * gamma[tid] + beta[tid];
  float y1 = d1 * rstd * gamma[tid + 256] + beta[tid + 256];
  xout[base + tid] = y0;
  xout[base + tid + 256] = y1;
  xh[base + tid] = (bf16)y0;
  xh[base + tid + 256] = (bf16)y1;
}

// ---------------------------------------------------------------------------
// Mean pool over sequence + tiny MLP head
// ---------------------------------------------------------------------------
__global__ void pool_kernel(const float* __restrict__ x, float* __restrict__ pooled) {
  int b = blockIdx.x;
  for (int c = threadIdx.x; c < D_; c += 256) {
    float s = 0.f;
    for (int t = 0; t < S_; ++t) s += x[((size_t)b * S_ + t) * D_ + c];
    pooled[b * D_ + c] = s * (1.f / (float)S_);
  }
}

__global__ void mlp1_kernel(const float* __restrict__ pooled, const float* __restrict__ w,
                            const float* __restrict__ bias, float* __restrict__ h) {
  int b = blockIdx.x;
  for (int f = threadIdx.x; f < FF_; f += 256) {
    float acc = bias[f];
    for (int d = 0; d < D_; ++d) acc += pooled[b * D_ + d] * w[(size_t)d * FF_ + f];
    h[b * FF_ + f] = fmaxf(acc, 0.f);
  }
}

__global__ void mlp2_kernel(const float* __restrict__ h, const float* __restrict__ w,
                            const float* __restrict__ bias, float* __restrict__ out) {
  int t = blockIdx.x * 256 + threadIdx.x;
  if (t < B_ * C_) {
    int b = t / C_, c = t % C_;
    float acc = bias[c];
    for (int f = 0; f < FF_; ++f) acc += h[(size_t)b * FF_ + f] * w[(size_t)f * C_ + c];
    out[t] = acc;
  }
}

// ---------------------------------------------------------------------------
// Launch
// ---------------------------------------------------------------------------
extern "C" void kernel_launch(void* const* d_in, const int* in_sizes, int n_in,
                              void* d_out, int out_size, void* d_ws, size_t ws_size,
                              hipStream_t stream) {
  const int*   tokens = (const int*)  d_in[0];
  const float* emb    = (const float*)d_in[1];
  const float* qkv_w  = (const float*)d_in[2];
  const float* qkv_b  = (const float*)d_in[3];
  const float* fc_w   = (const float*)d_in[4];
  const float* fc_b   = (const float*)d_in[5];
  const float* gamma  = (const float*)d_in[6];
  const float* beta   = (const float*)d_in[7];
  const float* fc1_w  = (const float*)d_in[8];
  const float* fc1_b  = (const float*)d_in[9];
  const float* fc2_w  = (const float*)d_in[10];
  const float* fc2_b  = (const float*)d_in[11];
  float* out = (float*)d_out;
  (void)in_sizes; (void)n_in; (void)out_size; (void)ws_size;

  char* ws = (char*)d_ws;
  size_t off = 0;
  auto take = [&](size_t bytes) -> char* {
    char* p = ws + off;
    off = (off + bytes + 255) & ~(size_t)255;
    return p;
  };
  float* x      = (float*)take((size_t)B_ * S_ * D_ * 4);
  bf16*  xh     = (bf16*) take((size_t)B_ * S_ * D_ * 2);
  bf16*  qkvbuf = (bf16*) take((size_t)B_ * S_ * QKV3 * 2);
  bf16*  obuf   = (bf16*) take((size_t)B_ * S_ * D_ * 2);
  float* proj   = (float*)take((size_t)B_ * S_ * D_ * 4);
  bf16*  qkvwT  = (bf16*) take((size_t)L_ * QKV3 * D_ * 2);
  bf16*  fcwT   = (bf16*) take((size_t)L_ * D_ * D_ * 2);
  float* pooled = (float*)take((size_t)B_ * D_ * 4);
  float* hbuf   = (float*)take((size_t)B_ * FF_ * 4);

  const int M = B_ * S_;
  convw_kernel<<<12288, 256, 0, stream>>>(qkv_w, fc_w, qkvwT, fcwT);
  embed_kernel<<<(M * D_) / 256, 256, 0, stream>>>(tokens, emb, x, xh);
  for (int l = 0; l < L_; ++l) {
    gemm_kernel<true><<<dim3(M / 128, QKV3 / 128), 256, 0, stream>>>(
        xh, qkvwT + (size_t)l * QKV3 * D_, qkv_b + l * QKV3, qkvbuf, M, QKV3, D_);
    attn_kernel<<<dim3(B_ * H_, S_ / 64), 128, 0, stream>>>(qkvbuf, obuf);
    gemm_kernel<false><<<dim3(M / 128, D_ / 128), 256, 0, stream>>>(
        obuf, fcwT + (size_t)l * D_ * D_, fc_b + l * D_, proj, M, D_, D_);
    ln_kernel<<<M, 256, 0, stream>>>(proj, x, gamma, beta, x, xh);
  }
  pool_kernel<<<B_, 256, 0, stream>>>(x, pooled);
  mlp1_kernel<<<B_, 256, 0, stream>>>(pooled, fc1_w, fc1_b, hbuf);
  mlp2_kernel<<<1, 256, 0, stream>>>(hbuf, fc2_w, fc2_b, out);
}
